// Encoder_36515811951214
// MI455X (gfx1250) — matrix operand order repaired and checked by reading, not compiled, not run on hardware
//
#include <hip/hip_runtime.h>

typedef __bf16 v16bf __attribute__((ext_vector_type(16)));
typedef __bf16 v8bf  __attribute__((ext_vector_type(8)));
typedef float  v8f   __attribute__((ext_vector_type(8)));

#define HDIM 512

// ---------------- degree / norm ----------------
__global__ void k_init_ones(float* __restrict__ d, int n) {
    int i = blockIdx.x * blockDim.x + threadIdx.x;
    if (i < n) d[i] = 1.0f;   // self loop contributes 1 to in-degree
}

__global__ void k_deg_accum(const long long* __restrict__ dst, float* __restrict__ deg, long long E) {
    long long i = (long long)blockIdx.x * blockDim.x + threadIdx.x;
    if (i < E) atomicAdd(&deg[dst[i]], 1.0f);
}

__global__ void k_finalize_dinv(float* __restrict__ d, int n) {
    int i = blockIdx.x * blockDim.x + threadIdx.x;
    if (i < n) d[i] = rsqrtf(fmaxf(d[i], 1.0f));
}

// ---------------- one-time precision conversion ----------------
__global__ void k_f32_to_bf16(const float* __restrict__ in, __bf16* __restrict__ out, size_t n8) {
    size_t i = (size_t)blockIdx.x * blockDim.x + threadIdx.x;
    if (i >= n8) return;
    float4 a = *(const float4*)(in + i * 8);
    float4 b = *(const float4*)(in + i * 8 + 4);
    v8bf o;
    o[0] = (__bf16)a.x; o[1] = (__bf16)a.y; o[2] = (__bf16)a.z; o[3] = (__bf16)a.w;
    o[4] = (__bf16)b.x; o[5] = (__bf16)b.y; o[6] = (__bf16)b.z; o[7] = (__bf16)b.w;
    *(v8bf*)(out + i * 8) = o;
}

// W[k][n] (512x512 f32) -> Wt[n][k] bf16 (one-time, tiny)
__global__ void k_w_transpose_bf16(const float* __restrict__ W, __bf16* __restrict__ Wt) {
    int idx = blockIdx.x * blockDim.x + threadIdx.x;
    int k = idx >> 9, n = idx & (HDIM - 1);
    Wt[(size_t)n * HDIM + k] = (__bf16)W[idx];
}

// ---------------- WMMA bf16 GEMM: C[M,512] = A[M,512] @ B[512,512] ----------------
// A: bf16 row-major; Bt: bf16 transposed weights Bt[n][k].
// Block tile 128x64, 8 waves; wave w owns rows w*16..+15 x 64 cols.
// K chunked by 64 (two 32-k WMMA steps); double-buffered LDS, one barrier/chunk.
// A staged with 8-group permutation [g0,g2,g1,g3] per 32-k block so every lane's
// 16-bf16 fragment is one contiguous 32B LDS read.
#define LDS_STRIDE 80   // 64 + 16 pad bf16; 160B row, 32B-aligned fragment reads
__global__ __launch_bounds__(256) void k_gemm_bf16(const __bf16* __restrict__ A,
                                                   const __bf16* __restrict__ Bt,
                                                   float* __restrict__ C, int M) {
    __shared__ alignas(32) __bf16 As[2][128][LDS_STRIDE];
    __shared__ alignas(32) __bf16 Bs[2][64][LDS_STRIDE];

    const int tid  = threadIdx.x;
    const int wave = tid >> 5;
    const int lane = tid & 31;
    const int hf   = lane >> 4;
    const int mrow = lane & 15;

    const int block_m = blockIdx.x * 128;
    const int block_n = blockIdx.y * 64;

    v8f acc[4] = {};
    uint4 aR[4], bR[2];

    auto loadA = [&](int k0) {
#pragma unroll
        for (int i = 0; i < 4; ++i) {
            int idx = tid + i * 256;
            int r = idx >> 3, j = idx & 7;
            int grow = block_m + r;
            uint4 v = make_uint4(0u, 0u, 0u, 0u);
            if (grow < M) v = *(const uint4*)(A + (size_t)grow * HDIM + k0 + j * 8);
            aR[i] = v;
        }
    };
    auto storeA = [&](int buf) {
#pragma unroll
        for (int i = 0; i < 4; ++i) {
            int idx = tid + i * 256;
            int r = idx >> 3, j = idx & 7;
            int b = j >> 2, g = j & 3;
            int p = ((g & 1) << 1) | (g >> 1);      // 0,2,1,3
            *(uint4*)&As[buf][r][b * 32 + p * 8] = aR[i];
        }
    };
    auto loadB = [&](int k0) {
#pragma unroll
        for (int i = 0; i < 2; ++i) {
            int idx = tid + i * 256;
            int n = idx >> 3, j = idx & 7;
            bR[i] = *(const uint4*)(Bt + (size_t)(block_n + n) * HDIM + k0 + j * 8);
        }
    };
    auto storeB = [&](int buf) {
#pragma unroll
        for (int i = 0; i < 2; ++i) {
            int idx = tid + i * 256;
            int n = idx >> 3, j = idx & 7;
            int b = j >> 2, g = j & 3;
            int p = ((g & 1) << 1) | (g >> 1);      // 0,2,1,3
            *(uint4*)&Bs[buf][n][b * 32 + p * 8] = bR[i];
        }
    };
    auto compute = [&](int buf) {
#pragma unroll
        for (int s = 0; s < 2; ++s) {
            v16bf af = *(const v16bf*)&As[buf][wave * 16 + mrow][s * 32 + hf * 16];
            v16bf bfr[4];
#pragma unroll
            for (int t = 0; t < 4; ++t)
                bfr[t] = *(const v16bf*)&Bs[buf][t * 16 + mrow][s * 32 + hf * 16];
#pragma unroll
            for (int t = 0; t < 4; ++t)
                acc[t] = __builtin_amdgcn_wmma_f32_16x16x32_bf16(
                    false, af, false, bfr[t], (short)0, acc[t], false, false);
        }
    };

    // prologue: stage chunk 0
    loadA(0); loadB(0);
    storeA(0); storeB(0);
    __syncthreads();

    int cur = 0;
    for (int k0 = 0; k0 < HDIM; k0 += 64) {
        const bool more = (k0 + 64) < HDIM;
        if (more) { loadA(k0 + 64); loadB(k0 + 64); }   // overlap with WMMAs below
        compute(cur);
        if (more) { storeA(cur ^ 1); storeB(cur ^ 1); } // waits loadcnt here, not earlier
        __syncthreads();
        cur ^= 1;
    }

    // D layout: VGPR r, lanes 0-15 -> (M=r, N=lane); lanes 16-31 -> (M=8+r, N=lane-16)
    const int row_base = block_m + wave * 16 + hf * 8;
    const int col0 = block_n + mrow;
#pragma unroll
    for (int t = 0; t < 4; ++t) {
#pragma unroll
        for (int r = 0; r < 8; ++r) {
            int row = row_base + r;
            if (row < M) C[(size_t)row * HDIM + col0 + t * 16] = acc[t][r];
        }
    }
}

// ---------------- aggregation ----------------
// out[i][j] = bias[j] + h[i][j] * dinv[i]^2   (self loop + bias init)
__global__ void k_agg_init(const float* __restrict__ h, const float* __restrict__ dinv,
                           const float* __restrict__ bias, float* __restrict__ out, int N) {
    size_t idx = (size_t)blockIdx.x * blockDim.x + threadIdx.x;
    if (idx >= (size_t)N * 128) return;
    int row = (int)(idx >> 7);
    int c4  = (int)(idx & 127) << 2;
    float d = dinv[row];
    float w = d * d;
    float4 hv = *(const float4*)(h + (size_t)row * HDIM + c4);
    float4 bv = *(const float4*)(bias + c4);
    float4 o;
    o.x = bv.x + hv.x * w; o.y = bv.y + hv.y * w;
    o.z = bv.z + hv.z * w; o.w = bv.w + hv.w * w;
    *(float4*)(out + (size_t)row * HDIM + c4) = o;
}

// scatter: out[dst] += h[src] * dinv[src]*dinv[dst].
// One wave per edge (metadata wave-uniform -> scalar loads); 4 edges per wave
// with software prefetch of the next edge's 2KB feature row (global_prefetch_b8).
#define EPW 4
__global__ __launch_bounds__(256) void k_agg_edges(const float* __restrict__ h,
                                                   const long long* __restrict__ src,
                                                   const long long* __restrict__ dst,
                                                   const float* __restrict__ dinv,
                                                   float* __restrict__ out, long long E) {
    long long ebase = ((long long)blockIdx.x * 8 + (threadIdx.x >> 5)) * EPW;
    int lane = threadIdx.x & 31;
    if (ebase < E)  // prefetch first row of this wave's batch
        __builtin_prefetch(h + (size_t)src[ebase] * HDIM + lane * 16, 0, 3);
#pragma unroll
    for (int q = 0; q < EPW; ++q) {
        long long e = ebase + q;
        if (e >= E) return;
        long long s = src[e], d = dst[e];
        if (q + 1 < EPW && e + 1 < E)   // prefetch next row while this one drains
            __builtin_prefetch(h + (size_t)src[e + 1] * HDIM + lane * 16, 0, 3);
        float w = dinv[s] * dinv[d];
        const float* hp = h + (size_t)s * HDIM;
        float* op = out + (size_t)d * HDIM;
#pragma unroll
        for (int i = 0; i < 4; ++i) {
            int c = (lane + i * 32) << 2;
            float4 hv = *(const float4*)(hp + c);
            atomicAdd(op + c + 0, hv.x * w);
            atomicAdd(op + c + 1, hv.y * w);
            atomicAdd(op + c + 2, hv.z * w);
            atomicAdd(op + c + 3, hv.w * w);
        }
    }
}

// ---------------- batch norm ----------------
__global__ void k_zero_stats(float* __restrict__ s) {
    int i = blockIdx.x * blockDim.x + threadIdx.x;
    if (i < 1024) s[i] = 0.0f;
}

__global__ void k_bn_stats(const float* __restrict__ X, float* __restrict__ sum,
                           float* __restrict__ sumsq, int N) {
    int col  = blockIdx.x * 64 + (threadIdx.x & 63);
    int rsub = threadIdx.x >> 6;
    int r0   = blockIdx.y * 512 + rsub;
    int rend = min(N, (int)(blockIdx.y + 1) * 512);
    float s = 0.f, sq = 0.f;
    for (int r = r0; r < rend; r += 4) {
        float v = X[(size_t)r * HDIM + col];
        s += v; sq += v * v;
    }
    atomicAdd(&sum[col], s);
    atomicAdd(&sumsq[col], sq);
}

// layer-1: normalize + ReLU, emit bf16 (only feeds the next GEMM)
__global__ void k_bn_apply_relu_bf16(const float* __restrict__ X, const float* __restrict__ sum,
                                     const float* __restrict__ sumsq, const float* __restrict__ gamma,
                                     const float* __restrict__ beta, __bf16* __restrict__ out, int N) {
    size_t idx = (size_t)blockIdx.x * blockDim.x + threadIdx.x;   // N*64 groups of 8
    if (idx >= (size_t)N * 64) return;
    int cb = (int)(idx & 63) << 3;
    size_t base = (idx >> 6) * HDIM + cb;
    float invN = 1.0f / (float)N;
    v8bf o;
#pragma unroll
    for (int i = 0; i < 8; ++i) {
        int col = cb + i;
        float mean = sum[col] * invN;
        float var  = sumsq[col] * invN - mean * mean;   // biased var (jnp.var)
        float v = (X[base + i] - mean) * rsqrtf(var + 1e-5f) * gamma[col] + beta[col];
        o[i] = (__bf16)fmaxf(v, 0.0f);
    }
    *(v8bf*)(out + base) = o;
}

// layer-2: normalize in place, f32 final output
__global__ void k_bn_apply_f32(float* __restrict__ X, const float* __restrict__ sum,
                               const float* __restrict__ sumsq, const float* __restrict__ gamma,
                               const float* __restrict__ beta, int N) {
    size_t idx = (size_t)blockIdx.x * blockDim.x + threadIdx.x;   // N*128 float4 slots
    if (idx >= (size_t)N * 128) return;
    int cb = (int)(idx & 127) << 2;
    size_t base = (idx >> 7) * HDIM + cb;
    float invN = 1.0f / (float)N;
    float4 xv = *(float4*)(X + base);
    float o[4] = {xv.x, xv.y, xv.z, xv.w};
#pragma unroll
    for (int i = 0; i < 4; ++i) {
        int col = cb + i;
        float mean = sum[col] * invN;
        float var  = sumsq[col] * invN - mean * mean;
        o[i] = (o[i] - mean) * rsqrtf(var + 1e-5f) * gamma[col] + beta[col];
    }
    float4 ov; ov.x = o[0]; ov.y = o[1]; ov.z = o[2]; ov.w = o[3];
    *(float4*)(X + base) = ov;
}

// ---------------- launch ----------------
extern "C" void kernel_launch(void* const* d_in, const int* in_sizes, int n_in,
                              void* d_out, int out_size, void* d_ws, size_t ws_size,
                              hipStream_t stream) {
    const float*     x   = (const float*)d_in[0];
    const long long* ei  = (const long long*)d_in[1];
    const float*     W1  = (const float*)d_in[2];
    const float*     b1  = (const float*)d_in[3];
    const float*     g1  = (const float*)d_in[4];
    const float*     be1 = (const float*)d_in[5];
    const float*     W2  = (const float*)d_in[6];
    const float*     b2  = (const float*)d_in[7];
    const float*     g2  = (const float*)d_in[8];
    const float*     be2 = (const float*)d_in[9];

    const int N = in_sizes[0] / HDIM;
    const long long E = in_sizes[1] / 2;
    const long long* src = ei;
    const long long* dst = ei + E;

    float* ws    = (float*)d_ws;
    int Npad     = (N + 63) & ~63;
    float* dinv  = ws;
    float* sums  = ws + Npad;             // 512 sum + 512 sumsq
    float* sumsq = sums + 512;
    float* tA    = sums + 1024;           // N*512 f32  (layer1 XW; reused as tC)
    float* tB    = tA + (size_t)N * HDIM; // N*512 f32  (aggregated features)
    __bf16* Xb   = (__bf16*)(tB + (size_t)N * HDIM);  // N*512 bf16 (reused as tBb)
    __bf16* W1t  = Xb + (size_t)N * HDIM;             // 512*512 bf16
    __bf16* W2t  = W1t + HDIM * HDIM;
    float* tC    = tA;        // alias: tA dead after layer-1 aggregation
    __bf16* tBb  = Xb;        // alias: Xb dead after GEMM1
    float* out   = (float*)d_out;

    const int TPB = 256;
    dim3 ggrid((N + 127) / 128, HDIM / 64);
    dim3 sgrid(HDIM / 64, (N + 511) / 512);
    size_t node4 = (size_t)N * 128;
    size_t node8 = (size_t)N * 64;
    unsigned egrid = (unsigned)((E + 8 * EPW - 1) / (8 * EPW));

    // symmetric norm from in-degree (incl. self loops)
    k_init_ones<<<(N + TPB - 1) / TPB, TPB, 0, stream>>>(dinv, N);
    k_deg_accum<<<(unsigned)((E + TPB - 1) / TPB), TPB, 0, stream>>>(dst, dinv, E);
    k_finalize_dinv<<<(N + TPB - 1) / TPB, TPB, 0, stream>>>(dinv, N);

    // one-time bf16 conversions for the matrix pipe
    k_f32_to_bf16<<<(unsigned)((node8 + TPB - 1) / TPB), TPB, 0, stream>>>(x, Xb, node8);
    k_w_transpose_bf16<<<HDIM * HDIM / TPB, TPB, 0, stream>>>(W1, W1t);
    k_w_transpose_bf16<<<HDIM * HDIM / TPB, TPB, 0, stream>>>(W2, W2t);

    // ---- layer 1: GEMM -> aggregate -> BN+ReLU (bf16 out) ----
    k_gemm_bf16<<<ggrid, TPB, 0, stream>>>(Xb, W1t, tA, N);
    k_agg_init<<<(unsigned)((node4 + TPB - 1) / TPB), TPB, 0, stream>>>(tA, dinv, b1, tB, N);
    k_agg_edges<<<egrid, TPB, 0, stream>>>(tA, src, dst, dinv, tB, E);
    k_zero_stats<<<4, TPB, 0, stream>>>(sums);
    k_bn_stats<<<sgrid, TPB, 0, stream>>>(tB, sums, sumsq, N);
    k_bn_apply_relu_bf16<<<(unsigned)((node8 + TPB - 1) / TPB), TPB, 0, stream>>>(
        tB, sums, sumsq, g1, be1, tBb, N);

    // ---- layer 2: GEMM -> aggregate -> BN ----
    k_gemm_bf16<<<ggrid, TPB, 0, stream>>>(tBb, W2t, tC, N);
    k_agg_init<<<(unsigned)((node4 + TPB - 1) / TPB), TPB, 0, stream>>>(tC, dinv, b2, out, N);
    k_agg_edges<<<egrid, TPB, 0, stream>>>(tC, src, dst, dinv, out, E);
    k_zero_stats<<<4, TPB, 0, stream>>>(sums);
    k_bn_stats<<<sgrid, TPB, 0, stream>>>(out, sums, sumsq, N);
    k_bn_apply_f32<<<(unsigned)((node4 + TPB - 1) / TPB), TPB, 0, stream>>>(
        out, sums, sumsq, g2, be2, N);
}
